// SelfAttentionV1_46832323396340
// MI455X (gfx1250) — matrix operand order repaired and checked by reading, not compiled, not run on hardware
//
#include <hip/hip_runtime.h>
#include <cstddef>

typedef __attribute__((ext_vector_type(16))) _Float16 v16h;
typedef __attribute__((ext_vector_type(8)))  _Float16 v8h;
typedef __attribute__((ext_vector_type(8)))  float    v8f;
typedef __attribute__((ext_vector_type(4)))  float    v4f;

#define B_ 4
#define S_ 4096
#define D_ 1024
#define SCALE_ 0.03125f   /* 1/sqrt(1024) */
#define KC_ 128           /* keys per LDS P chunk in pass 3 */
#define MT_ 4             /* row sub-tiles per wave in qkv pass */

union V16U { v16h v; v8h h[2]; };

__device__ __forceinline__ v8f wmma_f16(v16h a, v16h b, v8f c) {
  // D = A(16x32 f16) * B(32x16 f16) + C(16x16 f32)
  return __builtin_amdgcn_wmma_f32_16x16x32_f16(
      /*neg_a=*/false, a, /*neg_b=*/false, b,
      /*c_mod=*/(short)0, c, /*reuse_a=*/false, /*reuse_b=*/false);
}

__device__ __forceinline__ v8f vzero8() {
  v8f z = {0.f, 0.f, 0.f, 0.f, 0.f, 0.f, 0.f, 0.f};
  return z;
}

// Convert 8 contiguous fp32 -> 8 f16 (for A fragments built from fp32 x)
__device__ __forceinline__ v8h cvt8(const float* __restrict__ p) {
  const v4f f0 = *(const v4f*)p;
  const v4f f1 = *(const v4f*)(p + 4);
  v8h r;
  r[0] = (_Float16)f0[0]; r[1] = (_Float16)f0[1];
  r[2] = (_Float16)f0[2]; r[3] = (_Float16)f0[3];
  r[4] = (_Float16)f1[0]; r[5] = (_Float16)f1[1];
  r[6] = (_Float16)f1[2]; r[7] = (_Float16)f1[3];
  return r;
}

// ---------------------------------------------------------------- pass 0
__global__ void f32_to_f16_kernel(const float* __restrict__ src,
                                  _Float16* __restrict__ dst, int n) {
  int i = blockIdx.x * blockDim.x + threadIdx.x;
  if (i < n) dst[i] = (_Float16)src[i];
}

// ---------------------------------------------------------------- pass 1
// One wave per 64x16 output strip (MT_=4 row sub-tiles): the three weight
// B-fragments are loaded once per 32-k step and reused across 4 A-fragments
// -> 12 WMMAs per step, ~18 FLOP/B from L2.
__global__ __launch_bounds__(128)
void qkv_proj_kernel(const float* __restrict__ x,
                     const _Float16* __restrict__ Wqh,
                     const _Float16* __restrict__ Wkh,
                     const _Float16* __restrict__ Wvh,
                     const float* __restrict__ bq,
                     const float* __restrict__ bk,
                     const float* __restrict__ bv,
                     _Float16* __restrict__ Qh,
                     _Float16* __restrict__ Kh,
                     _Float16* __restrict__ Vt) {
  const int lane = threadIdx.x & 31;
  const int wave = threadIdx.x >> 5;
  const int tile = blockIdx.x * 4 + wave;   // (B*S/64)*(D/16) = 16384 strips
  const int rt0  = (tile >> 6) * MT_;       // first 16-row tile of strip
  const int ct   = tile & 63;               // col-tile over D (0..63)
  const int l15  = lane & 15;
  const int hi   = lane >> 4;

  const int row0 = rt0 * 16 + l15;          // A-fragment M index, sub-tile 0
  const int col  = ct * 16 + l15;           // B-fragment N index
  // A (16-bit, 16x32): per lane one 16-elem chunk at k + hi*16
  const float*    abase = x + (size_t)row0 * D_ + hi * 16;
  // B (16-bit, 32x16): per lane one 16-elem chunk at k + hi*16 (row `col` of W,
  // since Q = x * W^T makes W rows the K-major columns of B)
  const _Float16* wq = Wqh + (size_t)col * D_ + hi * 16;
  const _Float16* wk = Wkh + (size_t)col * D_ + hi * 16;
  const _Float16* wv = Wvh + (size_t)col * D_ + hi * 16;

  v8f cq[MT_], ck[MT_], cv[MT_];
#pragma unroll
  for (int mt = 0; mt < MT_; ++mt) { cq[mt] = vzero8(); ck[mt] = vzero8(); cv[mt] = vzero8(); }

  for (int k = 0; k < D_; k += 32) {
    const v16h fq = *(const v16h*)(wq + k);
    const v16h fk = *(const v16h*)(wk + k);
    const v16h fv = *(const v16h*)(wv + k);
#pragma unroll
    for (int mt = 0; mt < MT_; ++mt) {
      const float* ap = abase + (size_t)mt * 16 * D_ + k;
      V16U a;
      a.h[0] = cvt8(ap);
      a.h[1] = cvt8(ap + 8);
      cq[mt] = wmma_f16(a.v, fq, cq[mt]);
      ck[mt] = wmma_f16(a.v, fk, ck[mt]);
      cv[mt] = wmma_f16(a.v, fv, cv[mt]);
    }
  }

  const float biasq = bq[col], biask = bk[col], biasv = bv[col];
#pragma unroll
  for (int mt = 0; mt < MT_; ++mt) {
    const int rt = rt0 + mt;
    // C layout: lane<16 -> M = rt*16 + r ; lane>=16 -> M = rt*16 + 8 + r
    const int m0 = rt * 16 + hi * 8;
#pragma unroll
    for (int r = 0; r < 8; ++r) {
      const size_t mi = (size_t)(m0 + r);
      Qh[mi * D_ + col] = (_Float16)(cq[mt][r] + biasq);
      Kh[mi * D_ + col] = (_Float16)(ck[mt][r] + biask);
    }
    // V stored transposed: Vt[b][d][s] -> contiguous 16B store per lane
    const int bb = rt >> 8;                   // batch = rt / (S/16)
    const int s0 = (rt & 255) * 16 + hi * 8;  // position within batch
    v8h pv;
#pragma unroll
    for (int r = 0; r < 8; ++r) pv[r] = (_Float16)(cv[mt][r] + biasv);
    *(v8h*)(Vt + ((size_t)bb * D_ + col) * S_ + s0) = pv;
  }
}

// ---------------------------------------------------------------- pass 2
// One wave per 16-query tile: online softmax stats (row max m, denom l)
// over all keys, using S^T = K * Q^T tiles so reductions are per-lane.
// Two key tiles per iteration reuse each Q B-fragment across 2 WMMAs.
__global__ __launch_bounds__(128)
void attn_stats_kernel(const _Float16* __restrict__ Qh,
                       const _Float16* __restrict__ Kh,
                       float* __restrict__ mOut,
                       float* __restrict__ lOut) {
  const int lane  = threadIdx.x & 31;
  const int qt    = blockIdx.x * 4 + (threadIdx.x >> 5);  // 0..1023
  const int b     = qt >> 8;                               // S/16 = 256 tiles/batch
  const int qrow0 = qt * 16;                               // global row in [B*S]
  const int l15   = lane & 15;
  const int hi    = lane >> 4;

  const _Float16* bq = Qh + (size_t)(qrow0 + l15) * D_ + hi * 16;  // B frag (Q^T)
  float mrun = -3.402823466e+38f;
  float lrun = 0.0f;

  for (int kt = 0; kt < S_ / 16; kt += 2) {
    const _Float16* ak0 = Kh + (size_t)(b * S_ + kt * 16 + l15) * D_ + hi * 16;
    const _Float16* ak1 = ak0 + (size_t)16 * D_;
    // prefetch next pair of key tiles (streaming dimension)
    __builtin_prefetch(ak0 + (size_t)32 * D_, 0, 0);
    __builtin_prefetch(ak1 + (size_t)32 * D_, 0, 0);
    v8f c0 = vzero8(), c1 = vzero8();
    for (int d = 0; d < D_; d += 32) {
      const v16h fb = *(const v16h*)(bq + d);
      V16U a0, a1;
      a0.h[0] = *(const v8h*)(ak0 + d);
      a0.h[1] = *(const v8h*)(ak0 + d + 8);
      a1.h[0] = *(const v8h*)(ak1 + d);
      a1.h[1] = *(const v8h*)(ak1 + d + 8);
      c0 = wmma_f16(a0.v, fb, c0);
      c1 = wmma_f16(a1.v, fb, c1);
    }
    // lane holds 16 scores of query l15 (keys kt*16 + hi*8 + r, two tiles)
    float s0[8], s1[8];
    float mnew = mrun;
#pragma unroll
    for (int i = 0; i < 8; ++i) { s0[i] = c0[i] * SCALE_; mnew = fmaxf(mnew, s0[i]); }
#pragma unroll
    for (int i = 0; i < 8; ++i) { s1[i] = c1[i] * SCALE_; mnew = fmaxf(mnew, s1[i]); }
    float acc = lrun * __expf(mrun - mnew);
#pragma unroll
    for (int i = 0; i < 8; ++i) acc += __expf(s0[i] - mnew) + __expf(s1[i] - mnew);
    lrun = acc; mrun = mnew;
  }
  // combine the two key-halves held by lane q and lane q+16
  const float mo = __shfl_xor(mrun, 16, 32);
  const float lo = __shfl_xor(lrun, 16, 32);
  const float mt = fmaxf(mrun, mo);
  const float lt = lrun * __expf(mrun - mt) + lo * __expf(mo - mt);
  if (lane < 16) {
    mOut[qrow0 + l15] = mt;
    lOut[qrow0 + l15] = lt;
  }
}

// ---------------------------------------------------------------- pass 3
// 8-wave workgroup per 16-query tile. Per 128-key chunk:
//   stage A: each wave computes one S^T tile (Q from LDS), writes
//            normalized probs P[q][k] (f16) to LDS.
//   stage B: each wave accumulates O^T = V^T * P^T for its 8 d-tiles,
//            accumulators live in VGPRs across the whole key loop
//            (no rescale: m,l are exact from pass 2).
__global__ __launch_bounds__(256)
void attn_out_kernel(const _Float16* __restrict__ Qh,
                     const _Float16* __restrict__ Kh,
                     const _Float16* __restrict__ Vt,
                     const float* __restrict__ mIn,
                     const float* __restrict__ lIn,
                     float* __restrict__ out) {
  __shared__ __align__(32) _Float16 Qs[16 * D_];   // 32 KB
  __shared__ __align__(32) _Float16 Ps[16 * KC_];  // 4 KB
  const int tid   = threadIdx.x;
  const int lane  = tid & 31;
  const int w     = tid >> 5;          // wave id 0..7
  const int qt    = blockIdx.x;        // 0..1023
  const int b     = qt >> 8;
  const int qrow0 = qt * 16;
  const int l15   = lane & 15;
  const int hi    = lane >> 4;

  { // stage the 16xD Q tile into LDS once
    const v8h* src = (const v8h*)(Qh + (size_t)qrow0 * D_);
    v8h* dst = (v8h*)Qs;
    for (int i = tid; i < 16 * D_ / 8; i += 256) dst[i] = src[i];
  }
  const float mq = mIn[qrow0 + l15];
  const float rl = 1.0f / lIn[qrow0 + l15];

  v8f acc[8];
#pragma unroll
  for (int j = 0; j < 8; ++j) acc[j] = vzero8();

  __syncthreads();

  for (int kc0 = 0; kc0 < S_; kc0 += KC_) {
    // ---- stage A: this wave's score tile, keys kc0 + w*16 .. +15
    const _Float16* ak  = Kh + (size_t)(b * S_ + kc0 + w * 16 + l15) * D_ + hi * 16;
    const _Float16* bqs = Qs + (size_t)l15 * D_ + hi * 16;
    if (kc0 + KC_ < S_) {   // prefetch next chunk's key rows
      __builtin_prefetch(ak + (size_t)KC_ * D_, 0, 0);
    }
    v8f c = vzero8();
    for (int d = 0; d < D_; d += 32) {
      V16U a;
      a.h[0] = *(const v8h*)(ak + d);
      a.h[1] = *(const v8h*)(ak + d + 8);
      v16h fb = *(const v16h*)(bqs + d);
      c = wmma_f16(a.v, fb, c);
    }
    v8h p;
#pragma unroll
    for (int i = 0; i < 8; ++i)
      p[i] = (_Float16)(__expf(c[i] * SCALE_ - mq) * rl);
    // C layout of S^T: lane q holds keys (w*16 + hi*8 + 0..7) -> contiguous
    *(v8h*)(Ps + l15 * KC_ + w * 16 + hi * 8) = p;
    __syncthreads();

    // ---- stage B: O^T accumulation for my 8 d-tiles
#pragma unroll
    for (int j = 0; j < 8; ++j) {
      const int dt = w + j * 8;  // d-tile 0..63
      const _Float16* av = Vt + ((size_t)b * D_ + dt * 16 + l15) * S_ + kc0 + hi * 16;
      const _Float16* bp = Ps + l15 * KC_ + hi * 16;
      if (kc0 + KC_ < S_) {   // prefetch next chunk of V^T for this d-tile
        __builtin_prefetch(av + KC_, 0, 0);
      }
      for (int kk = 0; kk < KC_; kk += 32) {
        V16U a;
        a.h[0] = *(const v8h*)(av + kk);
        a.h[1] = *(const v8h*)(av + kk + 8);
        v16h fb = *(const v16h*)(bp + kk);
        acc[j] = wmma_f16(a.v, fb, acc[j]);
      }
    }
    __syncthreads();
  }

  // O^T[d][q] -> out[b][q][d]; lane writes 8 contiguous fp32 (32B)
#pragma unroll
  for (int j = 0; j < 8; ++j) {
    const int dt = w + j * 8;
    const int d0 = dt * 16 + hi * 8;
    float* o = out + (size_t)(qrow0 + l15) * D_ + d0;
    *(v8f*)o = acc[j];
  }
}

// ---------------------------------------------------------------- launch
extern "C" void kernel_launch(void* const* d_in, const int* in_sizes, int n_in,
                              void* d_out, int out_size, void* d_ws, size_t ws_size,
                              hipStream_t stream) {
  (void)in_sizes; (void)n_in; (void)out_size; (void)ws_size;
  const float* x  = (const float*)d_in[0];
  const float* Wq = (const float*)d_in[1];
  const float* bq = (const float*)d_in[2];
  const float* Wk = (const float*)d_in[3];
  const float* bk = (const float*)d_in[4];
  const float* Wv = (const float*)d_in[5];
  const float* bv = (const float*)d_in[6];
  float* out = (float*)d_out;

  // workspace carve-up (~102 MB)
  char* w = (char*)d_ws;
  _Float16* Wqh = (_Float16*)w; w += (size_t)D_ * D_ * 2;
  _Float16* Wkh = (_Float16*)w; w += (size_t)D_ * D_ * 2;
  _Float16* Wvh = (_Float16*)w; w += (size_t)D_ * D_ * 2;
  _Float16* Qh  = (_Float16*)w; w += (size_t)B_ * S_ * D_ * 2;
  _Float16* Kh  = (_Float16*)w; w += (size_t)B_ * S_ * D_ * 2;
  _Float16* Vt  = (_Float16*)w; w += (size_t)B_ * S_ * D_ * 2;
  float* mArr = (float*)w; w += (size_t)B_ * S_ * 4;
  float* lArr = (float*)w;

  const int nW = D_ * D_;
  f32_to_f16_kernel<<<(nW + 255) / 256, 256, 0, stream>>>(Wq, Wqh, nW);
  f32_to_f16_kernel<<<(nW + 255) / 256, 256, 0, stream>>>(Wk, Wkh, nW);
  f32_to_f16_kernel<<<(nW + 255) / 256, 256, 0, stream>>>(Wv, Wvh, nW);

  const int qkvStrips = (B_ * S_ / (16 * MT_)) * (D_ / 16);   // 16384
  qkv_proj_kernel<<<qkvStrips / 4, 128, 0, stream>>>(
      x, Wqh, Wkh, Wvh, bq, bk, bv, Qh, Kh, Vt);

  const int qTiles = B_ * S_ / 16;                   // 1024
  attn_stats_kernel<<<qTiles / 4, 128, 0, stream>>>(Qh, Kh, mArr, lArr);
  attn_out_kernel<<<qTiles, 256, 0, stream>>>(Qh, Kh, Vt, mArr, lArr, out);
}
